// MultiHeadSelfAttention_27745488732246
// MI455X (gfx1250) — matrix operand order repaired and checked by reading, not compiled, not run on hardware
//
#include <hip/hip_runtime.h>
#include <hip/hip_bf16.h>

// ---------------- problem constants ----------------
constexpr int B  = 4;
constexpr int T  = 2048;
constexpr int C  = 1024;
constexpr int H  = 16;
constexpr int HD = 64;

// ---------------- types ----------------
typedef __bf16 bf16;
typedef __attribute__((ext_vector_type(8)))  bf16  bf16x8;
typedef __attribute__((ext_vector_type(16))) bf16  bf16x16;
typedef __attribute__((ext_vector_type(8)))  float floatx8;

#define WAIT_ASYNC(n) asm volatile("s_wait_asynccnt " #n ::: "memory")
#define WAIT_DSCNT0   asm volatile("s_wait_dscnt 0" ::: "memory")

__device__ __forceinline__ bf16 f2bf(float f) {
  union { float f; unsigned u; } v; v.f = f;
  unsigned r = v.u + 0x7FFFu + ((v.u >> 16) & 1u);   // round-to-nearest-even
  unsigned short h = (unsigned short)(r >> 16);
  bf16 out; __builtin_memcpy(&out, &h, 2);
  return out;
}

// Async DMA: global -> LDS, 16 bytes per lane, tracked by ASYNCcnt.
// VDST carries the 32-bit LDS byte offset (low 32 bits of generic LDS ptr).
__device__ __forceinline__ void async_b128(void* lds_ptr, const void* gaddr) {
  unsigned loff = (unsigned)(size_t)lds_ptr;
  asm volatile("global_load_async_to_lds_b128 %0, %1, off"
               :: "v"(loff), "v"(gaddr) : "memory");
}

// A-fragment (16x32 bf16, M x K): lane l holds row (l&15);
// K = kb..kb+7 in v0..3, K = kb+16..kb+23 in v4..7, kb = 8*(l>>4).
__device__ __forceinline__ bf16x16 load_a_frag(const bf16* tile, int ld) {
  int l   = threadIdx.x & 31;
  int row = l & 15;
  int kb  = (l >> 4) << 3;
  const bf16* p = tile + row * ld + kb;
  bf16x16 f;
  ((bf16x8*)&f)[0] = *(const bf16x8*)(p);
  ((bf16x8*)&f)[1] = *(const bf16x8*)(p + 16);
  return f;
}

// B-fragment (32x16 bf16, K x N) from an N-major tile (tileT[n*ld + k]):
// lane l holds column (l&15); K = kb..kb+7 in v0..3, K = kb+16..kb+23 in v4..7, kb = 8*(l>>4).
__device__ __forceinline__ bf16x16 load_b_frag(const bf16* tileT, int ld) {
  int l   = threadIdx.x & 31;
  int col = l & 15;
  int kb  = (l >> 4) << 3;
  const bf16* p = tileT + col * ld + kb;
  bf16x16 f;
  ((bf16x8*)&f)[0] = *(const bf16x8*)(p);
  ((bf16x8*)&f)[1] = *(const bf16x8*)(p + 16);
  return f;
}

__device__ __forceinline__ floatx8 wmma_bf16(bf16x16 a, bf16x16 b, floatx8 c) {
  return __builtin_amdgcn_wmma_f32_16x16x32_bf16(false, a, false, b,
                                                 (short)0, c, false, false);
}

// ---------------- conversion / transpose ----------------
__global__ void cvt_bf16_kernel(const float* __restrict__ src,
                                bf16* __restrict__ dst, int n) {
  int i = blockIdx.x * 256 + threadIdx.x;
  if (i < n) dst[i] = f2bf(src[i]);
}

// src [z][rows][cols] (f32) -> dst [z][cols][rows] (bf16)
__global__ void transpose_bf16_kernel(const float* __restrict__ src,
                                      bf16* __restrict__ dst,
                                      int rows, int cols) {
  int m = blockIdx.z;
  int i = blockIdx.x * 256 + threadIdx.x;
  if (i < rows * cols) {
    int r = i / cols, c = i % cols;
    size_t base = (size_t)m * rows * cols;
    dst[base + (size_t)c * rows + r] = f2bf(src[base + i]);
  }
}

// ---------------- QKV projection (async-staged weight tiles) ----------------
// xb [B,T,C] bf16; wt [H,HD,C] bf16 (N-major); bias [H,HD] f32.
// out: [B,H,T,HD] bf16, or (transposeV) [B,H,HD,T] bf16.
__global__ __launch_bounds__(256)
void qkv_proj_kernel(const bf16* __restrict__ xb, const bf16* __restrict__ wt,
                     const float* __restrict__ bias, bf16* __restrict__ out,
                     float scale, int transposeV) {
  __shared__ bf16 ldsB[2][64 * 32];          // N-major weight tile, ld = 32

  int tid = threadIdx.x;
  int w   = tid >> 5;                        // wave 0..7
  int b   = blockIdx.z, h = blockIdx.y;
  int t0  = blockIdx.x * 128 + w * 16;
  const bf16* A  = xb + ((size_t)b * T + t0) * C;
  const bf16* Bt = wt + (size_t)h * HD * C;

  // cooperative issue of one 64x32 weight tile (256 lanes x 16B = 4KB)
  int bn = tid >> 2;                         // 0..63  (N row)
  int bk = (tid & 3) * 8;                    // elem offset in K
  auto issueB = [&](int k, int buf) {
    async_b128(&ldsB[buf][bn * 32 + bk], Bt + (size_t)bn * C + k + bk);
  };
  auto tileB = [&](int it, bf16x16 a, floatx8* acc) {
    const bf16* bt = &ldsB[it & 1][0];
#pragma unroll
    for (int c = 0; c < 4; ++c)
      acc[c] = wmma_bf16(a, load_b_frag(bt + c * 16 * 32, 32), acc[c]);
  };

  constexpr int NT = C / 32;
  issueB(0, 0);
  floatx8 acc[4] = {};
#pragma unroll 2
  for (int it = 0; it < NT - 1; ++it) {      // branch-free steady state
    issueB((it + 1) * 32, (it + 1) & 1);
    WAIT_ASYNC(1);
    __syncthreads();
    __builtin_prefetch(A + it * 32 + 256, 0, 1);
    bf16x16 a = load_a_frag(A + it * 32, C);
    tileB(it, a, acc);
    __syncthreads();
  }
  {                                          // peeled epilogue
    WAIT_ASYNC(0);
    __syncthreads();
    bf16x16 a = load_a_frag(A + (NT - 1) * 32, C);
    tileB(NT - 1, a, acc);
  }

  int hl = (tid >> 4) & 1;
  int nl = tid & 15;
  size_t bh = (size_t)b * H + h;
#pragma unroll
  for (int c = 0; c < 4; ++c) {
    int d = c * 16 + nl;
    float bs = bias[h * HD + d];
#pragma unroll
    for (int r = 0; r < 8; ++r) {
      int t = t0 + r + 8 * hl;
      float v = (acc[c][r] + bs) * scale;
      if (!transposeV)
        out[(bh * T + t) * HD + d] = f2bf(v);
      else
        out[(bh * HD + d) * T + t] = f2bf(v);
    }
  }
}

// ---------------- flash attention (causal, async K/V staging) ----------------
// Qb [B,H,T,HD] (pre-scaled by 1/sqrt(HD)), Kb [B,H,T,HD], Vt [B,H,HD,T].
// attnb [B,T,C] bf16 with C index = h*HD + d (head concat).
__global__ __launch_bounds__(256)
void flash_attn_kernel(const bf16* __restrict__ Qb, const bf16* __restrict__ Kb,
                       const bf16* __restrict__ Vt, bf16* __restrict__ attnb) {
  __shared__ bf16 ldsK[2][32 * 64];          // K tile [s][d], ld = 64
  __shared__ bf16 ldsV[2][64 * 32];          // V tile [d][s], ld = 32
  __shared__ bf16 lds_p[8][16][32];          // per-wave P tile (16x32)

  int tid = threadIdx.x;
  int w   = tid >> 5;
  int b   = blockIdx.z, h = blockIdx.y;
  int q0  = blockIdx.x * 128 + w * 16;       // this wave's 16 query rows
  int hl  = (tid >> 4) & 1;
  int nl  = tid & 15;
  size_t bh = (size_t)b * H + h;

  const bf16* qptr = Qb + (bh * T + q0) * HD;
  bf16x16 qf0 = load_a_frag(qptr, HD);       // d = 0..31
  bf16x16 qf1 = load_a_frag(qptr + 32, HD);  // d = 32..63

  // cooperative issue of one 32x64 K tile + 64x32 V(T) tile
  int kr = tid >> 3, kc = (tid & 7) * 8;     // K: 32 rows x 8 lanes
  int vr = tid >> 2, vc = (tid & 3) * 8;     // V: 64 rows x 4 lanes
  const bf16* kbase = Kb + bh * T * HD;
  const bf16* vbase = Vt + bh * HD * (size_t)T;
  auto issueKV = [&](int s0, int buf) {
    async_b128(&ldsK[buf][kr * 64 + kc], kbase + (size_t)(s0 + kr) * HD + kc);
    async_b128(&ldsV[buf][vr * 32 + vc], vbase + (size_t)vr * T + s0 + vc);
  };

  floatx8 O[4] = {};
  float mrow[8], lrow[8];
#pragma unroll
  for (int r = 0; r < 8; ++r) { mrow[r] = -1e30f; lrow[r] = 0.f; }

  auto tileQK = [&](int it) {
    int s0 = it * 32;
    if (s0 >= q0 + 16) return;               // outside this wave's causal range
    const bf16* kt = &ldsK[it & 1][0];
    const bf16* vt = &ldsV[it & 1][0];

    // ---- S = Q * K^T for a 16x32 score tile ----
    floatx8 S0 = {}, S1 = {};
    S0 = wmma_bf16(qf0, load_b_frag(kt, 64), S0);
    S0 = wmma_bf16(qf1, load_b_frag(kt + 32, 64), S0);
    S1 = wmma_bf16(qf0, load_b_frag(kt + 16 * 64, 64), S1);
    S1 = wmma_bf16(qf1, load_b_frag(kt + 16 * 64 + 32, 64), S1);

    // ---- causal mask on ragged diagonal tile ----
    if (s0 + 31 > q0) {
#pragma unroll
      for (int r = 0; r < 8; ++r) {
        int t = q0 + r + 8 * hl;
        if (s0 + nl > t)      S0[r] = -1e30f;
        if (s0 + 16 + nl > t) S1[r] = -1e30f;
      }
    }

    // ---- online softmax update (16-lane half owns one row set) ----
#pragma unroll
    for (int r = 0; r < 8; ++r) {
      float rm = fmaxf(S0[r], S1[r]);
      rm = fmaxf(rm, __shfl_xor(rm, 1, 32));
      rm = fmaxf(rm, __shfl_xor(rm, 2, 32));
      rm = fmaxf(rm, __shfl_xor(rm, 4, 32));
      rm = fmaxf(rm, __shfl_xor(rm, 8, 32));
      float mnew = fmaxf(mrow[r], rm);
      float fac  = __expf(mrow[r] - mnew);
      float p0   = __expf(S0[r] - mnew);
      float p1   = __expf(S1[r] - mnew);
      float ps   = p0 + p1;
      ps += __shfl_xor(ps, 1, 32);
      ps += __shfl_xor(ps, 2, 32);
      ps += __shfl_xor(ps, 4, 32);
      ps += __shfl_xor(ps, 8, 32);
      lrow[r] = lrow[r] * fac + ps;
      mrow[r] = mnew;
      O[0][r] *= fac; O[1][r] *= fac; O[2][r] *= fac; O[3][r] *= fac;
      int trow = r + 8 * hl;
      lds_p[w][trow][nl]      = f2bf(p0);
      lds_p[w][trow][16 + nl] = f2bf(p1);
    }

    WAIT_DSCNT0;                             // per-wave P repack RAW

    // ---- O += P * V ----
    bf16x16 pf = load_a_frag(&lds_p[w][0][0], 32);
#pragma unroll
    for (int c = 0; c < 4; ++c)
      O[c] = wmma_bf16(pf, load_b_frag(vt + c * 16 * 32, 32), O[c]);
  };

  int ntiles = blockIdx.x * 4 + 4;           // keys up to block diagonal (>= 4)
  issueKV(0, 0);
  for (int it = 0; it < ntiles - 1; ++it) {  // branch-free steady state
    issueKV((it + 1) * 32, (it + 1) & 1);
    WAIT_ASYNC(2);
    __syncthreads();
    tileQK(it);
    __syncthreads();
  }
  {                                          // peeled epilogue
    WAIT_ASYNC(0);
    __syncthreads();
    tileQK(ntiles - 1);
  }

  // ---- normalize and write head-concat output ----
#pragma unroll
  for (int r = 0; r < 8; ++r) {
    float inv = 1.0f / lrow[r];
    int t = q0 + r + 8 * hl;
    size_t rowoff = ((size_t)b * T + t) * C + h * HD;
#pragma unroll
    for (int c = 0; c < 4; ++c)
      attnb[rowoff + c * 16 + nl] = f2bf(O[c][r] * inv);
  }
}

// ---------------- output projection (async-staged weight tiles) ----------------
// ab [B*T, C] bf16; wot [C(out), C(in)] bf16 (N-major); bo [C] f32; out f32.
__global__ __launch_bounds__(256)
void out_proj_kernel(const bf16* __restrict__ ab, const bf16* __restrict__ wot,
                     const float* __restrict__ bo, float* __restrict__ out) {
  __shared__ bf16 ldsB[2][64 * 32];

  int tid = threadIdx.x;
  int w   = tid >> 5;
  int m0  = blockIdx.x * 128 + w * 16;       // rows over B*T
  int n0  = blockIdx.y * 64;
  const bf16* A  = ab  + (size_t)m0 * C;
  const bf16* Bt = wot + (size_t)n0 * C;

  int bn = tid >> 2;
  int bk = (tid & 3) * 8;
  auto issueB = [&](int k, int buf) {
    async_b128(&ldsB[buf][bn * 32 + bk], Bt + (size_t)bn * C + k + bk);
  };
  auto tileB = [&](int it, bf16x16 a, floatx8* acc) {
    const bf16* bt = &ldsB[it & 1][0];
#pragma unroll
    for (int c = 0; c < 4; ++c)
      acc[c] = wmma_bf16(a, load_b_frag(bt + c * 16 * 32, 32), acc[c]);
  };

  constexpr int NT = C / 32;
  issueB(0, 0);
  floatx8 acc[4] = {};
#pragma unroll 2
  for (int it = 0; it < NT - 1; ++it) {      // branch-free steady state
    issueB((it + 1) * 32, (it + 1) & 1);
    WAIT_ASYNC(1);
    __syncthreads();
    __builtin_prefetch(A + it * 32 + 256, 0, 1);
    bf16x16 a = load_a_frag(A + it * 32, C);
    tileB(it, a, acc);
    __syncthreads();
  }
  {                                          // peeled epilogue
    WAIT_ASYNC(0);
    __syncthreads();
    bf16x16 a = load_a_frag(A + (NT - 1) * 32, C);
    tileB(NT - 1, a, acc);
  }

  int hl = (tid >> 4) & 1;
  int nl = tid & 15;
#pragma unroll
  for (int c = 0; c < 4; ++c) {
    int n = n0 + c * 16 + nl;
    float bs = bo[n];
#pragma unroll
    for (int r = 0; r < 8; ++r) {
      int m = m0 + r + 8 * hl;
      out[(size_t)m * C + n] = acc[c][r] + bs;
    }
  }
}

// ---------------- host launch ----------------
extern "C" void kernel_launch(void* const* d_in, const int* in_sizes, int n_in,
                              void* d_out, int out_size, void* d_ws, size_t ws_size,
                              hipStream_t stream) {
  (void)in_sizes; (void)n_in; (void)out_size; (void)ws_size;
  const float* x  = (const float*)d_in[0];
  const float* Wq = (const float*)d_in[1];
  const float* bq = (const float*)d_in[2];
  const float* Wk = (const float*)d_in[3];
  const float* bk = (const float*)d_in[4];
  const float* Wv = (const float*)d_in[5];
  const float* bv = (const float*)d_in[6];
  const float* Wo = (const float*)d_in[7];
  const float* bo = (const float*)d_in[8];
  float* out = (float*)d_out;

  char* ws = (char*)d_ws;
  size_t off = 0;
  auto alloc = [&](size_t bytes) -> void* {
    void* p = ws + off;
    off = (off + bytes + 255) & ~(size_t)255;
    return p;
  };
  bf16* xb    = (bf16*)alloc((size_t)B * T * C * 2);      // [B,T,C]
  bf16* wqt   = (bf16*)alloc((size_t)H * C * HD * 2);     // [H,HD,C]
  bf16* wkt   = (bf16*)alloc((size_t)H * C * HD * 2);
  bf16* wvt   = (bf16*)alloc((size_t)H * C * HD * 2);
  bf16* wot   = (bf16*)alloc((size_t)C * C * 2);          // [Cout,Cin]
  bf16* Qb    = (bf16*)alloc((size_t)B * H * T * HD * 2); // [B,H,T,HD]
  bf16* Kb    = (bf16*)alloc((size_t)B * H * T * HD * 2);
  bf16* Vtb   = (bf16*)alloc((size_t)B * H * T * HD * 2); // [B,H,HD,T]
  bf16* attnb = (bf16*)alloc((size_t)B * T * C * 2);      // [B,T,C]

  // 1) convert x to bf16
  cvt_bf16_kernel<<<(B * T * C + 255) / 256, 256, 0, stream>>>(x, xb, B * T * C);

  // 2) transpose+convert weights to N-major bf16
  transpose_bf16_kernel<<<dim3((C * HD + 255) / 256, 1, H), 256, 0, stream>>>(Wq, wqt, C, HD);
  transpose_bf16_kernel<<<dim3((C * HD + 255) / 256, 1, H), 256, 0, stream>>>(Wk, wkt, C, HD);
  transpose_bf16_kernel<<<dim3((C * HD + 255) / 256, 1, H), 256, 0, stream>>>(Wv, wvt, C, HD);
  transpose_bf16_kernel<<<dim3((C * C + 255) / 256, 1, 1), 256, 0, stream>>>(Wo, wot, C, C);

  // 3) Q/K/V projections (Q pre-scaled by 1/sqrt(HD); V stored transposed)
  dim3 gqkv(T / 128, H, B);
  qkv_proj_kernel<<<gqkv, 256, 0, stream>>>(xb, wqt, bq, Qb, 0.125f, 0);
  qkv_proj_kernel<<<gqkv, 256, 0, stream>>>(xb, wkt, bk, Kb, 1.0f, 0);
  qkv_proj_kernel<<<gqkv, 256, 0, stream>>>(xb, wvt, bv, Vtb, 1.0f, 1);

  // 4) causal flash attention -> head-concat bf16
  flash_attn_kernel<<<dim3(T / 128, H, B), 256, 0, stream>>>(Qb, Kb, Vtb, attnb);

  // 5) output projection -> f32
  out_proj_kernel<<<dim3(B * T / 128, C / 64, 1), 256, 0, stream>>>(attnb, wot, bo, out);
}
